// PointNetPlusEncoder_73151882985681
// MI455X (gfx1250) — compile-verified, matrix operand order checked by reading, not run on hardware
//
#include <hip/hip_runtime.h>
#include <hip/hip_bf16.h>

typedef __attribute__((ext_vector_type(16))) _Float16 v16h;
typedef __attribute__((ext_vector_type(8)))  float    v8f;

#define BN_EPS 1e-5f

// ============================================================================
// Weight prep: f32 [Cout,Cin] -> f16 [Cout,Kpad] (zero padded K), fold BN into
// per-output-channel (scale, bias).
// ============================================================================
__global__ void prep_weights_kernel(const float* __restrict__ W, const float* __restrict__ bias,
                                    const float* __restrict__ g, const float* __restrict__ be,
                                    const float* __restrict__ m, const float* __restrict__ v,
                                    int Cin, int Kpad, int bnmode,
                                    _Float16* __restrict__ W16, float2* __restrict__ sb)
{
  const int o = blockIdx.x;
  const float* wrow = W + (size_t)o * Cin;
  _Float16* orow = W16 + (size_t)o * Kpad;
  for (int c = threadIdx.x; c < Kpad; c += blockDim.x)
    orow[c] = (c < Cin) ? (_Float16)wrow[c] : (_Float16)0.f;
  if (threadIdx.x == 0) {
    float s, t;
    if (bnmode) { s = g[o] * rsqrtf(v[o] + BN_EPS); t = s * (bias[o] - m[o]) + be[o]; }
    else        { s = 1.f; t = bias[o]; }
    sb[o] = make_float2(s, t);
  }
}

// ============================================================================
// Farthest point sampling: one workgroup per batch; running min-distance array
// lives in LDS, block-wide argmax via LDS tree reduction (first-max tiebreak,
// matching jnp.argmax).
// ============================================================================
__global__ __launch_bounds__(256)
void fps_kernel(const float* __restrict__ xyz, int N, int npoint,
                int* __restrict__ idx_out, float* __restrict__ newxyz)
{
  __shared__ float dist[4096];
  __shared__ float rv[256];
  __shared__ int   ri[256];
  const int b = blockIdx.x;
  const int t = threadIdx.x;
  const int T = blockDim.x;
  const float* xb = xyz + (size_t)b * N * 3;

  for (int n = t; n < N; n += T) dist[n] = 1e10f;
  __syncthreads();

  int far = 0;
  for (int p = 0; p < npoint; ++p) {
    if (t == 0) idx_out[b * npoint + p] = far;
    if (t < 3)  newxyz[((size_t)b * npoint + p) * 3 + t] = xb[(size_t)far * 3 + t];
    const float cx = xb[(size_t)far * 3 + 0];
    const float cy = xb[(size_t)far * 3 + 1];
    const float cz = xb[(size_t)far * 3 + 2];
    float bv = -1.f; int bi = 0;
    for (int n = t; n < N; n += T) {
      const float dx = xb[(size_t)n * 3 + 0] - cx;
      const float dy = xb[(size_t)n * 3 + 1] - cy;
      const float dz = xb[(size_t)n * 3 + 2] - cz;
      const float d  = dx * dx + dy * dy + dz * dz;
      const float nd = fminf(dist[n], d);
      dist[n] = nd;
      if (nd > bv) { bv = nd; bi = n; }
    }
    rv[t] = bv; ri[t] = bi;
    __syncthreads();
    for (int off = T >> 1; off > 0; off >>= 1) {
      if (t < off) {
        if (rv[t + off] > rv[t] || (rv[t + off] == rv[t] && ri[t + off] < ri[t])) {
          rv[t] = rv[t + off]; ri[t] = ri[t + off];
        }
      }
      __syncthreads();
    }
    far = ri[0];
    __syncthreads();
  }
}

// ============================================================================
// Ball query: one wave32 per query center. Ballot + popcount prefix appends
// in-radius indices in ascending order (== jnp.sort of masked indices),
// padding the tail with the first hit.
// ============================================================================
__global__ __launch_bounds__(128)
void ballquery_kernel(const float* __restrict__ xyz, const float* __restrict__ newxyz,
                      float r2, int nsample, int N, int S, int B,
                      int* __restrict__ gidx)
{
  const int lane = threadIdx.x & 31;
  const int w = blockIdx.x * (blockDim.x >> 5) + (threadIdx.x >> 5);
  if (w >= B * S) return;
  const int b = w / S, s = w % S;
  const float* c = newxyz + ((size_t)b * S + s) * 3;
  const float cx = c[0], cy = c[1], cz = c[2];
  const float* xb = xyz + (size_t)b * N * 3;
  int* out = gidx + ((size_t)b * S + s) * nsample;

  int count = 0;
  int first = -1;
  for (int base = 0; base < N; base += 32) {
    const int i = base + lane;
    bool pred = false;
    if (i < N) {
      const float dx = xb[(size_t)i * 3 + 0] - cx;
      const float dy = xb[(size_t)i * 3 + 1] - cy;
      const float dz = xb[(size_t)i * 3 + 2] - cz;
      pred = (dx * dx + dy * dy + dz * dz) <= r2;
    }
    const unsigned mask = (unsigned)__ballot(pred);
    if (mask) {
      if (first < 0) first = base + __ffs((int)mask) - 1;
      const int pos = count + __popc(mask & ((1u << lane) - 1u));
      if (pred && pos < nsample) out[pos] = i;
      count += __popc(mask);
      if (count >= nsample) break;
    }
  }
  if (first < 0) first = 0;
  for (int pos = count + lane; pos < nsample; pos += 32) out[pos] = first;
}

// ============================================================================
// Generic WMMA MLP layer:  out[R,Cout] = act( A[R,Kpad] @ W16[Cout,Kpad]^T * s + t )
// 4 waves / block; block covers a 64x64 macro-tile, each wave a 32x32 tile:
// 2 A-frags x 2 B-frags -> 4 accumulators per K chunk (8 b128 loads : 4 WMMA),
// doubling WMMA density vs 1-tile waves. Fragment striping per ISA 7.12.2.
// ============================================================================
__global__ __launch_bounds__(128)
void wmma_layer_kernel(const _Float16* __restrict__ A, const _Float16* __restrict__ W,
                       const float2* __restrict__ sb,
                       _Float16* __restrict__ out16, float* __restrict__ out32,
                       int R, int Kpad, int Cout, int relu)
{
  const int lane = threadIdx.x & 31;
  const int wave = threadIdx.x >> 5;
  const int rowBase = blockIdx.x * 64 + (wave & 1) * 32;   // 32 rows / wave
  const int colBase = blockIdx.y * 64 + (wave >> 1) * 32;  // 32 cols / wave
  if (colBase >= Cout) return;

  const int hl = lane >> 4;   // which 16-lane half of the wave
  const int lr = lane & 15;
  const _Float16* a0 = A + (size_t)(rowBase + lr) * Kpad;
  const _Float16* a1 = a0 + (size_t)16 * Kpad;
  const _Float16* w0 = W + (size_t)(colBase + lr) * Kpad;
  const _Float16* w1 = w0 + (size_t)16 * Kpad;

  v8f acc00 = {}, acc01 = {}, acc10 = {}, acc11 = {};
  for (int k0 = 0; k0 < Kpad; k0 += 32) {
    union { v16h v; uint4 q[2]; } af0, af1, bf0, bf1;
    af0.q[0] = *(const uint4*)(a0 + k0 + hl * 8);
    af0.q[1] = *(const uint4*)(a0 + k0 + 16 + hl * 8);
    af1.q[0] = *(const uint4*)(a1 + k0 + hl * 8);
    af1.q[1] = *(const uint4*)(a1 + k0 + 16 + hl * 8);
    bf0.q[0] = *(const uint4*)(w0 + k0 + hl * 8);
    bf0.q[1] = *(const uint4*)(w0 + k0 + 16 + hl * 8);
    bf1.q[0] = *(const uint4*)(w1 + k0 + hl * 8);
    bf1.q[1] = *(const uint4*)(w1 + k0 + 16 + hl * 8);
    if (k0 + 32 < Kpad) {
      __builtin_prefetch(a0 + k0 + 32, 0, 3);   // global_prefetch_b8
      __builtin_prefetch(a1 + k0 + 32, 0, 3);
      __builtin_prefetch(w0 + k0 + 32, 0, 3);
      __builtin_prefetch(w1 + k0 + 32, 0, 3);
    }
    acc00 = __builtin_amdgcn_wmma_f32_16x16x32_f16(false, af0.v, false, bf0.v,
                                                   (short)0, acc00, false, false);
    acc01 = __builtin_amdgcn_wmma_f32_16x16x32_f16(false, af0.v, false, bf1.v,
                                                   (short)0, acc01, false, false);
    acc10 = __builtin_amdgcn_wmma_f32_16x16x32_f16(false, af1.v, false, bf0.v,
                                                   (short)0, acc10, false, false);
    acc11 = __builtin_amdgcn_wmma_f32_16x16x32_f16(false, af1.v, false, bf1.v,
                                                   (short)0, acc11, false, false);
  }

  union { v8f v; float f[8]; } res[2][2];
  res[0][0].v = acc00; res[0][1].v = acc01;
  res[1][0].v = acc10; res[1][1].v = acc11;

  const int ocol0 = colBase + lr;
  const float2 st0 = sb[ocol0];
  const float2 st1 = sb[ocol0 + 16];
#pragma unroll
  for (int rt = 0; rt < 2; ++rt) {
#pragma unroll
    for (int i = 0; i < 8; ++i) {
      const int orow = rowBase + rt * 16 + i + hl * 8;  // C layout: VGPR i -> M = i + 8*(lane/16)
      if (orow >= R) continue;
      float v0 = res[rt][0].f[i] * st0.x + st0.y;
      float v1 = res[rt][1].f[i] * st1.x + st1.y;
      if (relu) { v0 = fmaxf(v0, 0.f); v1 = fmaxf(v1, 0.f); }
      if (out16) {
        out16[(size_t)orow * Cout + ocol0]      = (_Float16)v0;
        out16[(size_t)orow * Cout + ocol0 + 16] = (_Float16)v1;
      }
      if (out32) {
        out32[(size_t)orow * Cout + ocol0]      = v0;
        out32[(size_t)orow * Cout + ocol0 + 16] = v1;
      }
    }
  }
}

// ============================================================================
// Gathers (build padded f16 A matrices) + pooling / concat helpers
// ============================================================================
__global__ void gather_sa1_kernel(const float* __restrict__ xyz, const int* __restrict__ gidx,
                                  const float* __restrict__ newxyz, _Float16* __restrict__ A)
{
  const int r = blockIdx.x * blockDim.x + threadIdx.x;   // b*512*32 + s*32 + j
  if (r >= 16 * 512 * 32) return;
  const int s = (r >> 5) & 511;
  const int b = r >> 14;
  const int idx = gidx[r];
  const float* p = xyz + ((size_t)b * 4096 + idx) * 3;
  const float* c = newxyz + ((size_t)b * 512 + s) * 3;
  union { _Float16 h[32]; uint4 q[4]; } row;
#pragma unroll
  for (int i = 0; i < 32; ++i) row.h[i] = (_Float16)0.f;
  row.h[0] = (_Float16)(p[0] - c[0]);
  row.h[1] = (_Float16)(p[1] - c[1]);
  row.h[2] = (_Float16)(p[2] - c[2]);
  uint4* dst = (uint4*)(A + (size_t)r * 32);
  dst[0] = row.q[0]; dst[1] = row.q[1]; dst[2] = row.q[2]; dst[3] = row.q[3];
}

__global__ void gather_sa2_kernel(const float* __restrict__ xyz1, const int* __restrict__ gidx,
                                  const float* __restrict__ newxyz2, const float* __restrict__ l1pts,
                                  _Float16* __restrict__ A)
{
  const int r = blockIdx.x * blockDim.x + threadIdx.x;   // b*128*64 + s*64 + j
  if (r >= 16 * 128 * 64) return;
  const int s = (r >> 6) & 127;
  const int b = r >> 13;
  const int idx = gidx[r];
  const float* p = xyz1 + ((size_t)b * 512 + idx) * 3;
  const float* c = newxyz2 + ((size_t)b * 128 + s) * 3;
  _Float16* row = A + (size_t)r * 160;
  row[0] = (_Float16)(p[0] - c[0]);
  row[1] = (_Float16)(p[1] - c[1]);
  row[2] = (_Float16)(p[2] - c[2]);
  const float* f = l1pts + ((size_t)b * 512 + idx) * 128;
  for (int i = 0; i < 128; ++i) row[3 + i] = (_Float16)f[i];
  for (int i = 131; i < 160; ++i) row[i] = (_Float16)0.f;
}

__global__ void gather_sa3_kernel(const float* __restrict__ newxyz2, const float* __restrict__ l2pts,
                                  _Float16* __restrict__ A)
{
  const int r = blockIdx.x * blockDim.x + threadIdx.x;   // b*128 + s
  if (r >= 16 * 128) return;
  const float* c = newxyz2 + (size_t)r * 3;
  _Float16* row = A + (size_t)r * 288;
  row[0] = (_Float16)c[0]; row[1] = (_Float16)c[1]; row[2] = (_Float16)c[2];
  const float* f = l2pts + (size_t)r * 256;
  for (int i = 0; i < 256; ++i) row[3 + i] = (_Float16)f[i];
  for (int i = 259; i < 288; ++i) row[i] = (_Float16)0.f;
}

__global__ void maxpool_kernel(const _Float16* __restrict__ in, int G, int S, int C,
                               float* __restrict__ out)
{
  const int t = blockIdx.x * blockDim.x + threadIdx.x;
  if (t >= G * C) return;
  const int g = t / C, c = t % C;
  float m = -1e30f;
  const _Float16* p = in + ((size_t)g * S) * C + c;
  for (int s = 0; s < S; ++s) m = fmaxf(m, (float)p[(size_t)s * C]);
  out[(size_t)g * C + c] = m;
}

__global__ void mean_kernel(const float* __restrict__ l2pts, float* __restrict__ out)
{
  const int t = blockIdx.x * blockDim.x + threadIdx.x;   // b*256 + c
  if (t >= 16 * 256) return;
  const int b = t / 256, c = t % 256;
  float s = 0.f;
  for (int i = 0; i < 128; ++i) s += l2pts[((size_t)b * 128 + i) * 256 + c];
  out[t] = s * (1.f / 128.f);
}

__global__ void concat_fc_kernel(const float* __restrict__ l2pool, const float* __restrict__ l3,
                                 _Float16* __restrict__ A)
{
  const int t = blockIdx.x * blockDim.x + threadIdx.x;
  if (t >= 16 * 768) return;
  const int b = t / 768, c = t % 768;
  const float v = (c < 256) ? l2pool[b * 256 + c] : l3[b * 512 + (c - 256)];
  A[t] = (_Float16)v;
}

// ============================================================================
// Host orchestration
// ============================================================================
struct LayerDef { int wi, bi, gi, bei, mi, vi, Cout, Cin, Kpad, bnmode; };

extern "C" void kernel_launch(void* const* d_in, const int* in_sizes, int n_in,
                              void* d_out, int out_size, void* d_ws, size_t ws_size,
                              hipStream_t stream)
{
  (void)in_sizes; (void)n_in; (void)out_size; (void)ws_size;

  const float* xyz = (const float*)d_in[0];

  // Layer table: sa1[3], sa2[3], sa3[3], fc1, fc2 (d_in indices)
  const LayerDef L[11] = {
    { 1,  2,  3,  4,  5,  6,  64,   3,  32, 1},
    { 7,  8,  9, 10, 11, 12,  64,  64,  64, 1},
    {13, 14, 15, 16, 17, 18, 128,  64,  64, 1},
    {19, 20, 21, 22, 23, 24, 128, 131, 160, 1},
    {25, 26, 27, 28, 29, 30, 128, 128, 128, 1},
    {31, 32, 33, 34, 35, 36, 256, 128, 128, 1},
    {37, 38, 39, 40, 41, 42, 256, 259, 288, 1},
    {43, 44, 45, 46, 47, 48, 512, 256, 256, 1},
    {49, 50, 51, 52, 53, 54, 512, 512, 512, 1},
    {55, 56, 57, 58, 59, 60, 512, 768, 768, 1},   // fc1: g=bn1_g, be=bn1_b, m=bn1_m, v=bn1_v
    {61, 62, 62, 62, 62, 62, 512, 512, 512, 0},   // fc2: plain bias
  };

  // ---- workspace bump allocator ----
  char* ws = (char*)d_ws;
  size_t off = 0;
  auto alloc = [&](size_t bytes) -> void* {
    void* p = ws + off;
    off = (off + bytes + 255) & ~(size_t)255;
    return p;
  };

  _Float16* W16[11]; float2* SB[11];
  for (int i = 0; i < 11; ++i) {
    W16[i] = (_Float16*)alloc((size_t)L[i].Cout * L[i].Kpad * sizeof(_Float16));
    SB[i]  = (float2*)alloc((size_t)L[i].Cout * sizeof(float2));
  }
  int*   idx1    = (int*)alloc(16 * 512 * sizeof(int));
  float* newxyz1 = (float*)alloc((size_t)16 * 512 * 3 * sizeof(float));
  int*   gidx1   = (int*)alloc((size_t)16 * 512 * 32 * sizeof(int));
  int*   idx2    = (int*)alloc(16 * 128 * sizeof(int));
  float* newxyz2 = (float*)alloc((size_t)16 * 128 * 3 * sizeof(float));
  int*   gidx2   = (int*)alloc((size_t)16 * 128 * 64 * sizeof(int));
  float* l1pts   = (float*)alloc((size_t)16 * 512 * 128 * sizeof(float));
  float* l2pts   = (float*)alloc((size_t)16 * 128 * 256 * sizeof(float));
  float* l3f     = (float*)alloc((size_t)16 * 512 * sizeof(float));
  float* l2pool  = (float*)alloc((size_t)16 * 256 * sizeof(float));
  const int R1 = 16 * 512 * 32;      // 262144 rows
  const int R2 = 16 * 128 * 64;      // 131072 rows
  const int R3 = 16 * 128;           // 2048 rows
  _Float16* bufA = (_Float16*)alloc((size_t)R2 * 160 * sizeof(_Float16));  // >= R1*64 too
  _Float16* bufB = (_Float16*)alloc((size_t)R1 * 128 * sizeof(_Float16));  // >= R2*256 too

  // ---- weight prep ----
  for (int i = 0; i < 11; ++i) {
    prep_weights_kernel<<<dim3(L[i].Cout), dim3(64), 0, stream>>>(
        (const float*)d_in[L[i].wi], (const float*)d_in[L[i].bi],
        (const float*)d_in[L[i].gi], (const float*)d_in[L[i].bei],
        (const float*)d_in[L[i].mi], (const float*)d_in[L[i].vi],
        L[i].Cin, L[i].Kpad, L[i].bnmode, W16[i], SB[i]);
  }

  auto layer = [&](int li, const _Float16* A, _Float16* o16, float* o32, int R, int relu) {
    dim3 grid((R + 63) / 64, (L[li].Cout + 63) / 64);
    wmma_layer_kernel<<<grid, dim3(128), 0, stream>>>(A, W16[li], SB[li], o16, o32,
                                                      R, L[li].Kpad, L[li].Cout, relu);
  };

  // ---- SA1 ----
  fps_kernel<<<dim3(16), dim3(256), 0, stream>>>(xyz, 4096, 512, idx1, newxyz1);
  ballquery_kernel<<<dim3((16 * 512 + 3) / 4), dim3(128), 0, stream>>>(
      xyz, newxyz1, 0.04f, 32, 4096, 512, 16, gidx1);
  gather_sa1_kernel<<<dim3(R1 / 256), dim3(256), 0, stream>>>(xyz, gidx1, newxyz1, bufA);
  layer(0, bufA, bufB, nullptr, R1, 1);
  layer(1, bufB, bufA, nullptr, R1, 1);
  layer(2, bufA, bufB, nullptr, R1, 1);
  maxpool_kernel<<<dim3((16 * 512 * 128 + 255) / 256), dim3(256), 0, stream>>>(
      bufB, 16 * 512, 32, 128, l1pts);

  // ---- SA2 ----
  fps_kernel<<<dim3(16), dim3(256), 0, stream>>>(newxyz1, 512, 128, idx2, newxyz2);
  ballquery_kernel<<<dim3((16 * 128 + 3) / 4), dim3(128), 0, stream>>>(
      newxyz1, newxyz2, 0.16f, 64, 512, 128, 16, gidx2);
  gather_sa2_kernel<<<dim3((R2 + 255) / 256), dim3(256), 0, stream>>>(
      newxyz1, gidx2, newxyz2, l1pts, bufA);
  layer(3, bufA, bufB, nullptr, R2, 1);
  layer(4, bufB, bufA, nullptr, R2, 1);
  layer(5, bufA, bufB, nullptr, R2, 1);
  maxpool_kernel<<<dim3((16 * 128 * 256 + 255) / 256), dim3(256), 0, stream>>>(
      bufB, 16 * 128, 64, 256, l2pts);

  // ---- SA3 (group all) ----
  gather_sa3_kernel<<<dim3((R3 + 255) / 256), dim3(256), 0, stream>>>(newxyz2, l2pts, bufA);
  layer(6, bufA, bufB, nullptr, R3, 1);
  layer(7, bufB, bufA, nullptr, R3, 1);
  layer(8, bufA, bufB, nullptr, R3, 1);
  maxpool_kernel<<<dim3((16 * 512 + 255) / 256), dim3(256), 0, stream>>>(
      bufB, 16, 128, 512, l3f);

  // ---- head ----
  mean_kernel<<<dim3((16 * 256 + 255) / 256), dim3(256), 0, stream>>>(l2pts, l2pool);
  concat_fc_kernel<<<dim3((16 * 768 + 255) / 256), dim3(256), 0, stream>>>(l2pool, l3f, bufA);
  layer(9, bufA, bufB, nullptr, 16, 1);                 // fc1 + folded BN + relu
  layer(10, bufB, nullptr, (float*)d_out, 16, 0);       // fc2 -> f32 output [16,1,512]
}
